// SparseAttention_23149873725637
// MI455X (gfx1250) — matrix operand order repaired and checked by reading, not compiled, not run on hardware
//
#include <hip/hip_runtime.h>
#include <cstdint>
#include <cstddef>

// ---------------- problem constants ----------------
#define S_LEN      2048
#define D_MODEL    1024
#define N_HEADS    8
#define N_KVH      2
#define HEAD_DIM   128
#define QKV_Q      (N_HEADS * HEAD_DIM)   // 1024
#define QKV_KV     (N_KVH * HEAD_DIM)     // 256
#define LORA_R     64
#define GLOBAL_TOK 128
#define WIN_HALF_MAX 512                   // largest window / 2
#define ATTN_SCALE 0.08838834764831845f    // 128^-0.5

// ---------------- CDNA5 WMMA types ----------------
typedef __attribute__((ext_vector_type(16))) __bf16 v16bf;
typedef __attribute__((ext_vector_type(8)))  __bf16 v8bf;
typedef __attribute__((ext_vector_type(8)))  float  v8f;

__device__ __forceinline__ __bf16 f2bf(float f) {
  union { float f; uint32_t u; } v; v.f = f;
  uint32_t r = v.u + 0x7fffu + ((v.u >> 16) & 1u);   // round-to-nearest-even
  uint16_t h = (uint16_t)(r >> 16);
  return __builtin_bit_cast(__bf16, h);
}
__device__ __forceinline__ float bf2f(__bf16 b) {
  uint16_t h = __builtin_bit_cast(uint16_t, b);
  union { uint32_t u; float f; } v; v.u = ((uint32_t)h) << 16;
  return v.f;
}
__device__ __forceinline__ int iabs_i(int x) { return x < 0 ? -x : x; }

__device__ __forceinline__ v8f zero8() {
  v8f z;
#pragma unroll
  for (int i = 0; i < 8; ++i) z[i] = 0.f;
  return z;
}

__device__ __forceinline__ v8f wmma_bf16(v16bf a, v16bf b, v8f c) {
  // D(16x16,f32) = A(16x32,bf16) x B(32x16,bf16) + C
  return __builtin_amdgcn_wmma_f32_16x16x32_bf16(false, a, false, b, (short)0, c,
                                                 false, false);
}

// A fragment: 16x32 bf16 tile, row-major source (stride ld elements).
// Lane (l&15) = row M; lanes<16 hold K {0..7,16..23}, lanes>=16 hold K {8..15,24..31}.
__device__ __forceinline__ v16bf load_a_gl(const __bf16* base, int ld, int lane) {
  const int row = lane & 15;
  const int kb  = (lane >> 4) << 3;          // 0 or 8
  const __bf16* p = base + (size_t)row * ld + kb;
  v8bf lo = *(const v8bf*)p;                 // K: kb .. kb+7
  v8bf hi = *(const v8bf*)(p + 16);          // K: kb+16 .. kb+23
  v16bf r;
#pragma unroll
  for (int i = 0; i < 8; ++i) { r[i] = lo[i]; r[8 + i] = hi[i]; }
  return r;
}

// B fragment: 32x16 bf16 tile, source is Bt = B^T stored (N x K) row-major.
// Lane (l&15) = column N; lanes<16 hold K {0..7,16..23}, lanes>=16 hold K {8..15,24..31}.
__device__ __forceinline__ v16bf load_b_gl(const __bf16* baseT, int ld, int lane) {
  const int n  = lane & 15;
  const int kb = (lane >> 4) << 3;           // 0 or 8
  const __bf16* p = baseT + (size_t)n * ld + kb;
  v8bf lo = *(const v8bf*)p;
  v8bf hi = *(const v8bf*)(p + 16);
  v16bf r;
#pragma unroll
  for (int i = 0; i < 8; ++i) { r[i] = lo[i]; r[8 + i] = hi[i]; }
  return r;
}

// 16-lane half-group reductions (C-fragment rows live across a 16-lane group).
__device__ __forceinline__ float hmax16(float v) {
#pragma unroll
  for (int m = 8; m >= 1; m >>= 1) v = fmaxf(v, __shfl_xor(v, m, 32));
  return v;
}
__device__ __forceinline__ float hsum16(float v) {
#pragma unroll
  for (int m = 8; m >= 1; m >>= 1) v += __shfl_xor(v, m, 32);
  return v;
}

// ---------------- small prep kernels ----------------
__global__ void cvt_bf16_kernel(const float* __restrict__ X, __bf16* __restrict__ Y, int n) {
  int i = blockIdx.x * blockDim.x + threadIdx.x;
  if (i < n) Y[i] = f2bf(X[i]);
}

// W_eff = W + (b @ a) / R  , emitted directly in bf16
__global__ void fold_lora_kernel(const float* __restrict__ W, const float* __restrict__ Amat,
                                 const float* __restrict__ Bmat, __bf16* __restrict__ Weff,
                                 int OUT, int D) {
  int idx = blockIdx.x * blockDim.x + threadIdx.x;
  if (idx >= OUT * D) return;
  int o = idx / D, d = idx - o * D;
  float acc = 0.f;
#pragma unroll 8
  for (int r = 0; r < LORA_R; ++r) acc += Bmat[o * LORA_R + r] * Amat[r * D + d];
  Weff[idx] = f2bf(W[idx] + acc * (1.0f / LORA_R));
}

// RoPE in place on bf16 [S][heads*128]; optional scale folded into Q.
__global__ void rope_kernel(__bf16* __restrict__ X, const float* __restrict__ freqs,
                            const int* __restrict__ start_pos, int heads, float scl) {
  int idx = blockIdx.x * blockDim.x + threadIdx.x;
  int total = S_LEN * heads * (HEAD_DIM / 2);
  if (idx >= total) return;
  int s = idx / (heads * 64);
  int rem = idx - s * heads * 64;
  int h = rem / 64, i = rem - h * 64;
  int sp = start_pos[0];
  float c  = freqs[(size_t)(sp + s) * HEAD_DIM + i * 2 + 0];
  float sn = freqs[(size_t)(sp + s) * HEAD_DIM + i * 2 + 1];
  __bf16* p = X + (size_t)s * heads * HEAD_DIM + h * HEAD_DIM + i * 2;
  float x1 = bf2f(p[0]), x2 = bf2f(p[1]);
  p[0] = f2bf((x1 * c - x2 * sn) * scl);
  p[1] = f2bf((x1 * sn + x2 * c) * scl);
}

// ---------------- WMMA GEMM: C(MxN) = A(MxK,row) * Bt(NxK,row)^T ----------------
// MODE 0: f32 out, 1: bf16 out, 2: bf16 out transposed (C[col*ldc+row])
template <int MODE>
__global__ __launch_bounds__(256, 1)
void gemm_bf16_wmma(const __bf16* __restrict__ A, const __bf16* __restrict__ Bt,
                    void* __restrict__ C, int K, int lda, int ldb, int ldc) {
  const int lane = threadIdx.x & 31;
  const int wave = threadIdx.x >> 5;                 // 8 waves/block
  const int m0 = blockIdx.y * 128 + (wave >> 1) * 32; // wave: 32 rows
  const int n0 = blockIdx.x * 128 + (wave & 1) * 64;  // wave: 64 cols
  v8f acc[2][4];
#pragma unroll
  for (int i = 0; i < 2; ++i)
#pragma unroll
    for (int j = 0; j < 4; ++j) acc[i][j] = zero8();

  for (int k0 = 0; k0 < K; k0 += 32) {
    if (k0 + 32 < K) {  // global_prefetch_b8 of next K-tile
      __builtin_prefetch(A + (size_t)(m0 + (lane & 15)) * lda + k0 + 32, 0, 0);
      __builtin_prefetch(Bt + (size_t)(n0 + (lane & 15)) * ldb + k0 + 32, 0, 0);
    }
    v16bf a0 = load_a_gl(A + (size_t)m0 * lda + k0, lda, lane);
    v16bf a1 = load_a_gl(A + (size_t)(m0 + 16) * lda + k0, lda, lane);
#pragma unroll
    for (int nt = 0; nt < 4; ++nt) {
      v16bf b = load_b_gl(Bt + (size_t)(n0 + nt * 16) * ldb + k0, ldb, lane);
      acc[0][nt] = wmma_bf16(a0, b, acc[0][nt]);
      acc[1][nt] = wmma_bf16(a1, b, acc[1][nt]);
    }
  }
  const int nn = lane & 15;
  const int mb = (lane >> 4) << 3;
#pragma unroll
  for (int mi = 0; mi < 2; ++mi)
#pragma unroll
    for (int nt = 0; nt < 4; ++nt)
#pragma unroll
      for (int v = 0; v < 8; ++v) {
        int row = m0 + mi * 16 + mb + v;
        int col = n0 + nt * 16 + nn;
        float val = acc[mi][nt][v];
        if (MODE == 0)      ((float*)C)[(size_t)row * ldc + col] = val;
        else if (MODE == 1) ((__bf16*)C)[(size_t)row * ldc + col] = f2bf(val);
        else                ((__bf16*)C)[(size_t)col * ldc + row] = f2bf(val); // V^T
      }
}

// ---------------- fused triple-mask flash attention ----------------
// grid (S/64, H); 8 waves: wave = (q-subtile 0..3, dim-half 0..1)
// Each wave: 16 queries x 64 head-dims, 3 nested-mask accumulators,
// shared running max (valid since masks are nested), per-mask running sums.
__global__ __launch_bounds__(256, 1)
void attn_kernel(const __bf16* __restrict__ Qb,   // [S][1024], pre-scaled + roped
                 const __bf16* __restrict__ Kb,   // [S][256], roped
                 const __bf16* __restrict__ Vt,   // [256][S]  (V transposed)
                 const float*  __restrict__ scale_w,
                 __bf16* __restrict__ Ob) {       // [S][1024]
  __shared__ __bf16 lds_e[8][16 * 32];            // per-wave e-tile transpose staging
  const int lane = threadIdx.x & 31;
  const int wave = threadIdx.x >> 5;
  const int tq = wave >> 1;
  const int hf = wave & 1;
  const int q0 = blockIdx.x * 64 + tq * 16;
  const int h  = blockIdx.y;
  const int kh = h >> 2;                          // N_REP = 4

  const __bf16* Qh = Qb + (size_t)q0 * D_MODEL + h * HEAD_DIM;
  const __bf16* Kh = Kb + kh * HEAD_DIM;
  const __bf16* Vh = Vt + (size_t)kh * HEAD_DIM * S_LEN + (size_t)hf * 64 * S_LEN;

  v8f acc[3][4];
  float mrun[8], lrun[3][8];
#pragma unroll
  for (int w = 0; w < 3; ++w)
#pragma unroll
    for (int nt = 0; nt < 4; ++nt) acc[w][nt] = zero8();
#pragma unroll
  for (int v = 0; v < 8; ++v) {
    mrun[v] = -1e30f;
    lrun[0][v] = lrun[1][v] = lrun[2][v] = 0.f;
  }

  const int nn = lane & 15;
  const int mb = (lane >> 4) << 3;
  __bf16* myL = lds_e[wave];

  auto tile = [&](int k0) {
    // ---- scores: Q(16x128) x K^T tile (128x32) => two 16x16 C-fragments ----
    v8f s0 = zero8(), s1 = zero8();
#pragma unroll
    for (int kk = 0; kk < 4; ++kk) {
      v16bf aq = load_a_gl(Qh + kk * 32, D_MODEL, lane);
      v16bf b0 = load_b_gl(Kh + (size_t)k0 * QKV_KV + kk * 32, QKV_KV, lane);
      v16bf b1 = load_b_gl(Kh + (size_t)(k0 + 16) * QKV_KV + kk * 32, QKV_KV, lane);
      s0 = wmma_bf16(aq, b0, s0);
      s1 = wmma_bf16(aq, b1, s1);
    }
    // ---- online softmax stats (shared max, per-mask sums) ----
    float sf[8];
#pragma unroll
    for (int v = 0; v < 8; ++v) {
      const int q  = q0 + mb + v;
      const int kA = k0 + nn, kB = k0 + 16 + nn;
      const int dA = iabs_i(kA - q), dB = iabs_i(kB - q);
      const bool gA = kA < GLOBAL_TOK, gB = kB < GLOBAL_TOK;
      const bool a3A = (dA <= 512) || gA;
      const bool a3B = (dB <= 512) || gB;
      float x0 = s0[v], x1 = s1[v];
      float tm = hmax16(fmaxf(a3A ? x0 : -1e30f, a3B ? x1 : -1e30f));
      float mo = mrun[v];
      float mn = fmaxf(mo, tm);
      mrun[v] = mn;
      float scl = __expf(mo - mn);
      sf[v] = scl;
      float e0 = a3A ? __expf(x0 - mn) : 0.f;
      float e1 = a3B ? __expf(x1 - mn) : 0.f;
      float p1 = (((dA <= 128) || gA) ? e0 : 0.f) + (((dB <= 128) || gB) ? e1 : 0.f);
      float p2 = (((dA <= 256) || gA) ? e0 : 0.f) + (((dB <= 256) || gB) ? e1 : 0.f);
      lrun[0][v] = lrun[0][v] * scl + hsum16(p1);
      lrun[1][v] = lrun[1][v] * scl + hsum16(p2);
      lrun[2][v] = lrun[2][v] * scl + hsum16(e0 + e1);
      myL[(mb + v) * 32 + nn]      = f2bf(e0);   // C-layout -> LDS (16x32 tile)
      myL[(mb + v) * 32 + 16 + nn] = f2bf(e1);
    }
#pragma unroll
    for (int w = 0; w < 3; ++w)
#pragma unroll
      for (int nt = 0; nt < 4; ++nt)
#pragma unroll
        for (int v = 0; v < 8; ++v) acc[w][nt][v] *= sf[v];

    // ---- LDS transpose -> A-fragment; nested masks re-applied in-register ----
    const int arow = lane & 15;
    const int akb  = (lane >> 4) << 3;
    v8bf lo = *(const v8bf*)&myL[arow * 32 + akb];
    v8bf hi = *(const v8bf*)&myL[arow * 32 + akb + 16];
    v16bf ea3;
#pragma unroll
    for (int i = 0; i < 8; ++i) { ea3[i] = lo[i]; ea3[8 + i] = hi[i]; }
    v16bf ea1 = ea3, ea2 = ea3;
    const __bf16 zb = f2bf(0.f);
    const int qA = q0 + arow;
#pragma unroll
    for (int i = 0; i < 16; ++i) {
      int kloc = (i < 8) ? (akb + i) : (akb + 8 + i);   // element -> K index
      int k = k0 + kloc;
      int d = iabs_i(k - qA);
      bool g = k < GLOBAL_TOK;
      if (!((d <= 128) || g)) ea1[i] = zb;
      if (!((d <= 256) || g)) ea2[i] = zb;
    }
    // ---- AV: P(16x32) x V(32x64) per mask ----
#pragma unroll
    for (int nt = 0; nt < 4; ++nt) {
      v16bf bv = load_b_gl(Vh + (size_t)(nt * 16) * S_LEN + k0, S_LEN, lane);
      acc[0][nt] = wmma_bf16(ea1, bv, acc[0][nt]);
      acc[1][nt] = wmma_bf16(ea2, bv, acc[1][nt]);
      acc[2][nt] = wmma_bf16(ea3, bv, acc[2][nt]);
    }
  };

  // global columns [0,128) are in every mask; window band [q0-512, q0+15+512]
  for (int k0 = 0; k0 < GLOBAL_TOK; k0 += 32) tile(k0);
  int wlo = q0 - WIN_HALF_MAX;
  if (wlo < GLOBAL_TOK) wlo = GLOBAL_TOK;
  wlo &= ~31;
  if (wlo < GLOBAL_TOK) wlo = GLOBAL_TOK;
  int whi = q0 + 16 + WIN_HALF_MAX;
  whi = (whi + 31) & ~31;
  if (whi > S_LEN) whi = S_LEN;
  for (int k0 = wlo; k0 < whi; k0 += 32) tile(k0);

  // ---- epilogue: blend the three masked softmaxes ----
  float w0s = scale_w[0], w1s = scale_w[1], w2s = scale_w[2];
  float mw = fmaxf(w0s, fmaxf(w1s, w2s));
  float ew0 = __expf(w0s - mw), ew1 = __expf(w1s - mw), ew2 = __expf(w2s - mw);
  float inv = 1.f / (ew0 + ew1 + ew2);
  ew0 *= inv; ew1 *= inv; ew2 *= inv;
#pragma unroll
  for (int nt = 0; nt < 4; ++nt)
#pragma unroll
    for (int v = 0; v < 8; ++v) {
      float o = ew0 * acc[0][nt][v] / lrun[0][v]
              + ew1 * acc[1][nt][v] / lrun[1][v]
              + ew2 * acc[2][nt][v] / lrun[2][v];
      Ob[(size_t)(q0 + mb + v) * D_MODEL + h * HEAD_DIM + hf * 64 + nt * 16 + nn] =
          f2bf(o);
    }
}

// ---------------- host launcher ----------------
extern "C" void kernel_launch(void* const* d_in, const int* in_sizes, int n_in,
                              void* d_out, int out_size, void* d_ws, size_t ws_size,
                              hipStream_t stream) {
  (void)in_sizes; (void)n_in; (void)out_size; (void)ws_size;
  const float* x    = (const float*)d_in[0];
  const float* fc   = (const float*)d_in[1];
  const float* wq_w = (const float*)d_in[2];
  const float* wq_a = (const float*)d_in[3];
  const float* wq_b = (const float*)d_in[4];
  const float* wk_w = (const float*)d_in[5];
  const float* wk_a = (const float*)d_in[6];
  const float* wk_b = (const float*)d_in[7];
  const float* wv_w = (const float*)d_in[8];
  const float* wv_a = (const float*)d_in[9];
  const float* wv_b = (const float*)d_in[10];
  const float* wo_w = (const float*)d_in[11];
  const float* wo_a = (const float*)d_in[12];
  const float* wo_b = (const float*)d_in[13];
  const float* sw   = (const float*)d_in[14];
  const int*   sp   = (const int*)d_in[15];

  char* ws = (char*)d_ws;
  size_t off = 0;
  auto walloc = [&](size_t bytes) -> void* {
    void* p = ws + off;
    off = (off + bytes + 255) & ~(size_t)255;
    return p;
  };
  __bf16* xb  = (__bf16*)walloc((size_t)S_LEN * D_MODEL * 2);
  __bf16* wqe = (__bf16*)walloc((size_t)QKV_Q * D_MODEL * 2);
  __bf16* wke = (__bf16*)walloc((size_t)QKV_KV * D_MODEL * 2);
  __bf16* wve = (__bf16*)walloc((size_t)QKV_KV * D_MODEL * 2);
  __bf16* woe = (__bf16*)walloc((size_t)D_MODEL * QKV_Q * 2);
  __bf16* Qb  = (__bf16*)walloc((size_t)S_LEN * QKV_Q * 2);
  __bf16* Kb  = (__bf16*)walloc((size_t)S_LEN * QKV_KV * 2);
  __bf16* Vt  = (__bf16*)walloc((size_t)QKV_KV * S_LEN * 2);
  __bf16* Ob  = (__bf16*)walloc((size_t)S_LEN * QKV_Q * 2);

  dim3 blk(256);
  cvt_bf16_kernel<<<(S_LEN * D_MODEL) / 256, blk, 0, stream>>>(x, xb, S_LEN * D_MODEL);
  fold_lora_kernel<<<(QKV_Q * D_MODEL) / 256, blk, 0, stream>>>(wq_w, wq_a, wq_b, wqe, QKV_Q, D_MODEL);
  fold_lora_kernel<<<(QKV_KV * D_MODEL) / 256, blk, 0, stream>>>(wk_w, wk_a, wk_b, wke, QKV_KV, D_MODEL);
  fold_lora_kernel<<<(QKV_KV * D_MODEL) / 256, blk, 0, stream>>>(wv_w, wv_a, wv_b, wve, QKV_KV, D_MODEL);
  fold_lora_kernel<<<(D_MODEL * QKV_Q) / 256, blk, 0, stream>>>(wo_w, wo_a, wo_b, woe, D_MODEL, QKV_Q);

  // Q = x @ wq_eff^T   (bf16 out)
  gemm_bf16_wmma<1><<<dim3(QKV_Q / 128, S_LEN / 128), blk, 0, stream>>>(
      xb, wqe, Qb, D_MODEL, D_MODEL, D_MODEL, QKV_Q);
  // K = x @ wk_eff^T   (bf16 out)
  gemm_bf16_wmma<1><<<dim3(QKV_KV / 128, S_LEN / 128), blk, 0, stream>>>(
      xb, wke, Kb, D_MODEL, D_MODEL, D_MODEL, QKV_KV);
  // V^T = (x @ wv_eff^T)^T  (bf16, transposed store for WMMA B-fragments)
  gemm_bf16_wmma<2><<<dim3(QKV_KV / 128, S_LEN / 128), blk, 0, stream>>>(
      xb, wve, Vt, D_MODEL, D_MODEL, D_MODEL, S_LEN);

  rope_kernel<<<(S_LEN * N_HEADS * 64) / 256, blk, 0, stream>>>(Qb, fc, sp, N_HEADS, ATTN_SCALE);
  rope_kernel<<<(S_LEN * N_KVH * 64) / 256, blk, 0, stream>>>(Kb, fc, sp, N_KVH, 1.0f);

  attn_kernel<<<dim3(S_LEN / 64, N_HEADS), blk, 0, stream>>>(Qb, Kb, Vt, sw, Ob);

  // out = attn @ wo_eff^T  (f32 out to d_out)
  gemm_bf16_wmma<0><<<dim3(D_MODEL / 128, S_LEN / 128), blk, 0, stream>>>(
      Ob, woe, (float*)d_out, QKV_Q, QKV_Q, QKV_Q, D_MODEL);
}
